// SupConLoss_adv_61460982005779
// MI455X (gfx1250) — matrix operand order repaired and checked by reading, not compiled, not run on hardware
//
#include <hip/hip_runtime.h>
#include <math.h>

typedef __attribute__((ext_vector_type(16))) __bf16 v16bf;
typedef __attribute__((ext_vector_type(8)))  float  v8f;

#define NTOT    8192      // bsz * n_views
#define BSZ     4096
#define DDIM    256
#define JSPLIT  4         // column-loop partitions (fills more WGPs)
#define JT_TOT  (NTOT/16) // 512 column tiles
#define JT_PER  (JT_TOT/JSPLIT)
#define ROWS_PER_WG 128   // 8 waves x 16 rows
#define TEMP    0.07f
#define INVT    (1.0f/TEMP)
#define EPI     0.1f
#define BSTRIDE 272       // LDS row stride in bf16 elems (544B: 32B-aligned, banks spread)

// contrast row i = features[i % bsz, i / bsz, :]  (view-major concat)
__device__ __forceinline__ size_t crow_off(int i) {
    return ((size_t)(i & (BSZ - 1)) * 2 + (size_t)(i >> 12)) * DDIM;
}

__global__ void __launch_bounds__(256)
supcon_tiles(const float* __restrict__ feat, const int* __restrict__ labels,
             float* __restrict__ part /* [NTOT][JSPLIT][4] = m,Z,P,n */)
{
    __shared__ __bf16 Bhi[16 * BSTRIDE];
    __shared__ __bf16 Blo[16 * BSTRIDE];
    __shared__ float  sqArr[8][16];

    const int t    = threadIdx.x;
    const int wave = t >> 5;
    const int lane = t & 31;
    const int lo16 = lane & 15;
    const int hi   = lane >> 4;

    const int i0 = blockIdx.x * ROWS_PER_WG + wave * 16;

    // ---- A fragments: 16 rows x 256 K, bf16 hi + f32-residual lo (register resident)
    // A-layout (16x32 bf16): lane m<16 holds K {0..15}, lane m+16 holds {16..31}
    v16bf ahi[8], alo[8];
    float sqp = 0.f;
    {
        const float* ap = feat + crow_off(i0 + lo16);
#pragma unroll
        for (int kc = 0; kc < 8; ++kc) {
            const int k0 = kc * 32 + hi * 16;
#pragma unroll
            for (int g = 0; g < 2; ++g) {
                const float4 x0 = *(const float4*)(ap + k0 + g * 8);
                const float4 x1 = *(const float4*)(ap + k0 + g * 8 + 4);
                const float f[8] = {x0.x, x0.y, x0.z, x0.w, x1.x, x1.y, x1.z, x1.w};
#pragma unroll
                for (int e = 0; e < 8; ++e) {
                    const float v = f[e];
                    const __bf16 h = (__bf16)v;
                    ahi[kc][g * 8 + e] = h;
                    alo[kc][g * 8 + e] = (__bf16)(v - (float)h);
                    sqp += v * v;
                }
            }
        }
    }
    // lane m and m+16 loaded disjoint K halves of the same row -> combine for ||a||^2
    sqp += __shfl_xor(sqp, 16, 32);
    if (hi == 0) sqArr[wave][lo16] = sqp;
    __syncthreads();

    float sqv[8];
    int   rlab[8];
    const int rg0 = i0 + hi * 8;   // C-layout: VGPR v holds row v (lanes 0-15) / 8+v (16-31)
#pragma unroll
    for (int v = 0; v < 8; ++v) {
        sqv[v]  = sqArr[wave][hi * 8 + v];
        rlab[v] = labels[(rg0 + v) & (BSZ - 1)];
    }

    // online softmax state per lane: this lane sees column j0+lo16 of every tile
    float m[8], Z[8], P[8], n[8];
#pragma unroll
    for (int v = 0; v < 8; ++v) { m[v] = -__builtin_inff(); Z[v] = 0.f; P[v] = 0.f; n[v] = 0.f; }

    const int jt0 = blockIdx.y * JT_PER;
    for (int jt = jt0; jt < jt0 + JT_PER; ++jt) {
        const int j0 = jt * 16;
        // ---- cooperative convert-load of B tile (16 cols x 256 K) into LDS
        {
            const int brow = t >> 4;             // tile column index
            const int kseg = (t & 15) * 16;      // 16 floats per thread
            const float* bp = feat + crow_off(j0 + brow) + kseg;
            v16bf vh, vl;
#pragma unroll
            for (int q = 0; q < 4; ++q) {
                const float4 x = *(const float4*)(bp + q * 4);
                const float f[4] = {x.x, x.y, x.z, x.w};
#pragma unroll
                for (int e = 0; e < 4; ++e) {
                    const __bf16 h = (__bf16)f[e];
                    vh[q * 4 + e] = h;
                    vl[q * 4 + e] = (__bf16)(f[e] - (float)h);
                }
            }
            *(v16bf*)&Bhi[brow * BSTRIDE + kseg] = vh;
            *(v16bf*)&Blo[brow * BSTRIDE + kseg] = vl;
        }
        __syncthreads();

        const int clab = labels[(j0 + lo16) & (BSZ - 1)];

        // ---- 16x16 tile: K=256 in 8 chunks, 3 compensated bf16 WMMAs each
        v8f acc = {};
#pragma unroll
        for (int kc = 0; kc < 8; ++kc) {
            // B-layout used here (32x16, same K order as A): lane n holds col n, 16 consecutive K per half-wave
            const v16bf bh = *(const v16bf*)&Bhi[lo16 * BSTRIDE + kc * 32 + hi * 16];
            const v16bf bl = *(const v16bf*)&Blo[lo16 * BSTRIDE + kc * 32 + hi * 16];
            acc = __builtin_amdgcn_wmma_f32_16x16x32_bf16(false, ahi[kc], false, bh, (short)0, acc, false, false);
            acc = __builtin_amdgcn_wmma_f32_16x16x32_bf16(false, alo[kc], false, bh, (short)0, acc, false, false);
            acc = __builtin_amdgcn_wmma_f32_16x16x32_bf16(false, ahi[kc], false, bl, (short)0, acc, false, false);
        }

        // ---- epilogue: adversarial term, mask, online max/sum updates
        const int cg = j0 + lo16;
#pragma unroll
        for (int v = 0; v < 8; ++v) {
            const int  rg    = rg0 + v;
            const bool match = (clab == rlab[v]);
            const bool diag  = (cg == rg);
            const float raw  = (acc[v] - (match ? EPI * sqv[v] : 0.f)) * INVT;
            const float mn   = fmaxf(m[v], raw);            // diagonal included in max
            const float zs   = __expf(m[v] - mn);
            const float tm   = diag ? 0.f : __expf(raw - mn); // logits_mask excludes diag
            Z[v] = Z[v] * zs + tm;
            m[v] = mn;
            if (match && !diag) { P[v] += raw; n[v] += 1.f; }
        }
        __syncthreads();   // protect LDS before next tile overwrite
    }

    // ---- merge the 16 lanes holding the same rows (xor stays within half-wave)
#pragma unroll
    for (int off = 1; off <= 8; off <<= 1) {
#pragma unroll
        for (int v = 0; v < 8; ++v) {
            const float om = __shfl_xor(m[v], off, 32);
            const float oz = __shfl_xor(Z[v], off, 32);
            const float op = __shfl_xor(P[v], off, 32);
            const float on = __shfl_xor(n[v], off, 32);
            const float mn = fmaxf(m[v], om);
            Z[v] = Z[v] * __expf(m[v] - mn) + oz * __expf(om - mn);
            m[v] = mn; P[v] += op; n[v] += on;
        }
    }

    if (lo16 == 0) {
#pragma unroll
        for (int v = 0; v < 8; ++v) {
            float* q = part + ((size_t)(rg0 + v) * JSPLIT + blockIdx.y) * 4;
            q[0] = m[v]; q[1] = Z[v]; q[2] = P[v]; q[3] = n[v];
        }
    }
}

// merge JSPLIT partials per row, compute per-row loss, deterministic block sum
__global__ void __launch_bounds__(256)
supcon_rows(const float* __restrict__ part, float* __restrict__ blocksum)
{
    __shared__ float red[256];
    const int row = blockIdx.x * 256 + threadIdx.x;
    float m = -__builtin_inff(), Z = 0.f, P = 0.f, n = 0.f;
    for (int c = 0; c < JSPLIT; ++c) {
        const float* q = part + ((size_t)row * JSPLIT + c) * 4;
        const float mc = q[0], zc = q[1], pc = q[2], nc = q[3];
        const float mn = fmaxf(m, mc);
        Z = Z * __expf(m - mn) + zc * __expf(mc - mn);
        m = mn; P += pc; n += nc;
    }
    // mean_log_prob_pos = (P - n*(m + log Z)) / n ; loss = -(T/baseT)*that, T==baseT
    const float loss = -(P - n * (m + logf(Z))) / n;
    red[threadIdx.x] = loss;
    __syncthreads();
    for (int s = 128; s > 0; s >>= 1) {
        if (threadIdx.x < s) red[threadIdx.x] += red[threadIdx.x + s];
        __syncthreads();
    }
    if (threadIdx.x == 0) blocksum[blockIdx.x] = red[0];
}

__global__ void supcon_final(const float* __restrict__ blocksum, float* __restrict__ out)
{
    float v = blocksum[threadIdx.x];              // 32 block sums, one wave
    for (int off = 16; off > 0; off >>= 1) v += __shfl_xor(v, off, 32);
    if (threadIdx.x == 0) out[0] = v / (float)NTOT;
}

extern "C" void kernel_launch(void* const* d_in, const int* in_sizes, int n_in,
                              void* d_out, int out_size, void* d_ws, size_t ws_size,
                              hipStream_t stream) {
    const float* feat   = (const float*)d_in[0];   // [4096, 2, 256] f32
    const int*   labels = (const int*)d_in[1];     // [4096] int32
    float* out = (float*)d_out;

    float* part     = (float*)d_ws;                        // NTOT*JSPLIT*4 floats (512KB)
    float* blocksum = part + (size_t)NTOT * JSPLIT * 4;    // 32 floats

    dim3 grid1(NTOT / ROWS_PER_WG, JSPLIT);                // 64 x 4 = 256 WGs
    supcon_tiles<<<grid1, 256, 0, stream>>>(feat, labels, part);
    supcon_rows <<<NTOT / 256, 256, 0, stream>>>(part, blocksum);
    supcon_final<<<1, 32, 0, stream>>>(blocksum, out);
}
